// CppnPotentialCAStep_50818053046653
// MI455X (gfx1250) — matrix operand order repaired and checked by reading, compiled, not run on hardware
//
#include <hip/hip_runtime.h>

typedef __attribute__((ext_vector_type(16))) _Float16 v16h;
typedef __attribute__((ext_vector_type(8)))  _Float16 v8h;
typedef __attribute__((ext_vector_type(8)))  float    v8f;

#define SZ   96
#define CH   8
#define NP   16
#define TAPS 125          // 5*5*5
#define KPAD 1024         // 125*8 = 1000 -> pad to 1024 (zero B columns)
#define NKC  (KPAD / 32)  // 32 WMMA k-steps per tile
#define TX   16           // x positions per wave tile (WMMA M)
#define WY   2            // y rows per block tile
#define WZ   4            // z slabs per block tile (8 waves = WY*WZ)
#define NXT  (SZ / TX)    // 6
#define NYT  (SZ / WY)    // 48
#define NZT  (SZ / WZ)    // 24
#define SY   (WY + 4)     // 6 staged y rows
#define SX   (TX + 4)     // 20 staged x cols
#define SZS  (WZ + 5)     // 8 staged z slabs + 1 zero pad slab (taps 125..127)
#define NTILES  (4 * NXT * NYT * NZT)  // 27648 block-tiles
#define PERSIST 8                      // tiles per persistent block
#define NBLOCKS (NTILES / PERSIST)     // 3456 blocks

// ---- Prep: build the channel-embedded sparse GEMM B matrix once (32 KB in d_ws) ----
// BTg[p][k], k = tap*8 + c: kern[tap][p] if c == c0[p] (and tap < 125), else 0.
__global__ __launch_bounds__(256) void prep_bt_kernel(
    const float* __restrict__ kern, const int* __restrict__ c0,
    _Float16* __restrict__ BTg)
{
    const int i = blockIdx.x * 256 + threadIdx.x;
    if (i >= NP * KPAD) return;
    const int p = i >> 10, k = i & (KPAD - 1);
    const int tap = k >> 3, c = k & 7;
    float v = 0.0f;
    if (tap < TAPS && c == c0[p]) v = kern[tap * NP + p];
    BTg[i] = (_Float16)v;
}

__global__ __launch_bounds__(256) void ca_step_kernel(
    const float* __restrict__ in,   const _Float16* __restrict__ BTg,
    const float* __restrict__ mArr, const float* __restrict__ sArr,
    const float* __restrict__ hArr, const float* __restrict__ tArr,
    const int*   __restrict__ c1,   float* __restrict__ outp)
{
    // LDS: 32KB weights + 17.3KB staged tile + 0.5KB tap table + 4KB fields = 54.7KB
    __shared__ __align__(32) _Float16 BT[NP][KPAD];
    __shared__ __align__(16) _Float16 stg[SZS][SY][SX][CH];
    __shared__ int2  tapOff2[NKC * 2];
    __shared__ float fld[8][TX * CH];

    const int tid  = threadIdx.x;
    const int wave = tid >> 5, lane = tid & 31;
    const int hi   = lane >> 4;      // lane half (A/B fragment K-group select)
    const int col  = lane & 15;      // A row (x position) / B column (pair p)
    const int wy   = wave & 1, wz = wave >> 1;

    // ---- Loop-invariant gate parameters (one pair p per lane via 'col') ----
    const float mp  = mArr[col], sp = sArr[col], hp = hArr[col];
    const int   c1p = c1[col];
    const float inv2s2 = 0.5f / (sp * sp);
    const float invT   = 1.0f / tArr[0];

    // ---- Copy prebuilt B matrix into LDS: 8x (global_load_b128 + ds_store_b128) ----
    for (int i = tid; i < NP * KPAD / 8; i += 256)
        ((v8h*)&BT[0][0])[i] = ((const v8h*)BTg)[i];

    // ---- A-tap LDS byte-offset table: delta(tap) = ((dz*SY+dy)*SX+dx)*CH*2 ----
    for (int i = tid; i < NKC * 2; i += 256) {
        const int kc = i >> 1, half = i & 1;
        int d[2];
        #pragma unroll
        for (int g = 0; g < 2; ++g) {
            const int tap = kc * 4 + half * 2 + g;
            const int dz = tap / 25;
            const int rr = tap - dz * 25;
            const int dy = rr / 5;
            const int dx = rr - dy * 5;
            d[g] = (((dz * SY + dy) * SX) + dx) * (CH * 2);
        }
        tapOff2[i] = make_int2(d[0], d[1]);
    }

    // ---- Zero the pad slab once (staging never writes lz == SZS-1) ----
    for (int i = tid; i < SY * SX; i += 256) {
        v8h zz = {};
        *(v8h*)&stg[SZS - 1][i / SX][i % SX][0] = zz;
    }

    const char* sbl = (const char*)&stg[0][0][0][0]
                    + (((wz * SY + wy) * SX) + col) * (CH * 2);   // lane base byte offset

    // ---- Persistent loop: each block processes PERSIST consecutive block-tiles ----
    #pragma unroll 1
    for (int it = 0; it < PERSIST; ++it) {
        int t = blockIdx.x * PERSIST + it;
        const int xt = t % NXT; t /= NXT;
        const int yt = t % NYT; t /= NYT;
        const int zt = t % NZT; t /= NZT;
        const int bb = t;
        const int x0 = xt * TX, y0 = yt * WY, z0 = zt * WZ;

        __syncthreads();   // prior-iteration LDS readers done; BT/table visible on it==0

        // ---- Stage haloed input tile (circular wrap), f32 -> f16 ----
        for (int v = tid; v < (WZ + 4) * SY * SX; v += 256) {
            const int lz = v / (SY * SX);
            const int r  = v % (SY * SX);
            const int ly = r / SX, lx = r % SX;
            const int gz = (z0 + lz + SZ - 2) % SZ;
            const int gy = (y0 + ly + SZ - 2) % SZ;
            const int gx = (x0 + lx + SZ - 2) % SZ;
            const float* src = in + ((((long)bb * SZ + gz) * SZ + gy) * SZ + gx) * CH;
            const float4 lo  = *(const float4*)src;
            const float4 hi4 = *(const float4*)(src + 4);
            v8h d;
            d[0] = (_Float16)lo.x;  d[1] = (_Float16)lo.y;
            d[2] = (_Float16)lo.z;  d[3] = (_Float16)lo.w;
            d[4] = (_Float16)hi4.x; d[5] = (_Float16)hi4.y;
            d[6] = (_Float16)hi4.z; d[7] = (_Float16)hi4.w;
            *(v8h*)&stg[lz][ly][lx][0] = d;
        }
        __syncthreads();

        // ---- Per-wave GEMM tile: M=16 x-positions, N=16 pairs, K=1024 ----
        v8f acc = {};
        #pragma unroll
        for (int kc = 0; kc < NKC; ++kc) {
            const int2 dd = tapOff2[kc * 2 + hi];      // ds_load_b64: both tap deltas
            const v8h a0 = *(const v8h*)(sbl + dd.x);  // ds_load_b128: tap0, 8 channels
            const v8h a1 = *(const v8h*)(sbl + dd.y);  // ds_load_b128: tap1, 8 channels
            v16h A;
            #pragma unroll
            for (int e = 0; e < 8; ++e) { A[e] = a0[e]; A[8 + e] = a1[e]; }
            const v16h Bf = *(const v16h*)&BT[col][kc * 32 + hi * 16];
            acc = __builtin_amdgcn_wmma_f32_16x16x32_f16(
                false, A, false, Bf, (short)0, acc, false, false);
        }

        // ---- Gaussian gate + scatter-add into destination channels ----
        float* f = fld[wave];
        f[lane * 4 + 0] = 0.0f; f[lane * 4 + 1] = 0.0f;
        f[lane * 4 + 2] = 0.0f; f[lane * 4 + 3] = 0.0f;

        #pragma unroll
        for (int r = 0; r < 8; ++r) {
            const float pot = acc[r];             // C/D layout: M = r + 8*hi, N = col
            const float dmm = pot - mp;
            const float dlt = (__expf(-dmm * dmm * inv2s2) * 2.0f - 1.0f) * hp;
            atomicAdd(&f[(r + 8 * hi) * CH + c1p], dlt);   // ds_add_f32, wave-private
        }
        __syncthreads();

        // ---- out = clip(in + field/T, 0, 1); each lane writes 4 contiguous floats ----
        const int Mpos = lane >> 1, cb = (lane & 1) * 4;
        const int gy = y0 + wy, gz = z0 + wz, gx = x0 + Mpos;
        const long gi = ((((long)bb * SZ + gz) * SZ + gy) * SZ + gx) * CH + cb;
        const float4 iv = *(const float4*)(in + gi);
        const float4 fv = *(const float4*)&f[Mpos * CH + cb];
        float4 ov;
        ov.x = fminf(fmaxf(iv.x + fv.x * invT, 0.0f), 1.0f);
        ov.y = fminf(fmaxf(iv.y + fv.y * invT, 0.0f), 1.0f);
        ov.z = fminf(fmaxf(iv.z + fv.z * invT, 0.0f), 1.0f);
        ov.w = fminf(fmaxf(iv.w + fv.w * invT, 0.0f), 1.0f);
        *(float4*)(outp + gi) = ov;
    }
}

extern "C" void kernel_launch(void* const* d_in, const int* in_sizes, int n_in,
                              void* d_out, int out_size, void* d_ws, size_t ws_size,
                              hipStream_t stream) {
    (void)in_sizes; (void)n_in; (void)out_size; (void)ws_size;
    const float* in   = (const float*)d_in[0];
    const float* kern = (const float*)d_in[1];
    const float* m    = (const float*)d_in[2];
    const float* s    = (const float*)d_in[3];
    const float* h    = (const float*)d_in[4];
    const float* T    = (const float*)d_in[5];
    const int*   c0   = (const int*)d_in[6];
    const int*   c1   = (const int*)d_in[7];
    float* out        = (float*)d_out;
    _Float16* BTg     = (_Float16*)d_ws;   // 32 KB scratch for prebuilt B matrix

    prep_bt_kernel<<<(NP * KPAD + 255) / 256, 256, 0, stream>>>(kern, c0, BTg);
    ca_step_kernel<<<NBLOCKS, 256, 0, stream>>>(in, BTg, m, s, h, T, c1, out);
}
